// OriginalDH_SFNN_86474871538235
// MI455X (gfx1250) — matrix operand order repaired and checked by reading, not compiled, not run on hardware
//
#include <hip/hip_runtime.h>

// ---------------------------------------------------------------------------
// Fused block-masked GEMM + SNN (DH-SFNN) update for MI455X (gfx1250).
//
// Preferred path (ws_size >= 48MB):
//   pass 1: convert input (32MB f32) and W (64MB f32) -> bf16 in d_ws (once).
//   pass 2: GEMM on v_wmma_f32_16x16x32_bf16. Tiles staged to LDS with
//           global_load_async_to_lds_b128 (ASYNCcnt), 4-slot / 3-tiles-in-
//           flight pipeline so DMA of tiles kt+1..kt+3 overlaps WMMAs of kt.
//           (async loads complete in order, so s_wait_asynccnt 6 == oldest
//            of 3 in-flight tiles has landed)
// Fallback path: single fused kernel converting f32->bf16 while staging.
// 192MB L2 holds the whole input+W working set, so inter-block tile re-reads
// never touch HBM.
// ---------------------------------------------------------------------------

typedef __attribute__((ext_vector_type(16))) __bf16 v16bf;
typedef __attribute__((ext_vector_type(2)))  __bf16 v2bf;
typedef __attribute__((ext_vector_type(8)))  float  v8f;

#define B_DIM   2048          // batch
#define D_DIM   2048          // OUTPUT_DIM
#define K_DIM   4096          // INPUT_DIM (= W row stride)
#define BM      128           // batch rows per block
#define BN      64            // neurons per block
#define BK      32            // K per WMMA step
#define NKT     ((K_DIM / 2) / BK)        // 64 K-steps per phase
#define LDA     40            // padded LDS row (halves): 80B = 20 banks -> conflict free
#define LDB     40
#define ABUF    (BM * LDA)                // halves: A region inside a slot (5120)
#define TBUF    (BM * LDA + BN * LDB)     // halves per slot (7680)
#define NBUF    4                         // pipeline depth (slots)
#define SLOT_B  (TBUF * 2)                // bytes per slot (15360)

__device__ __forceinline__ float sigmoid_f(float x) {
  return 1.0f / (1.0f + __expf(-x));
}

// native bf16 pack
__device__ __forceinline__ unsigned pk2bf(float a, float b) {
  union { v2bf v; unsigned u; } c;
  c.v.x = (__bf16)a;
  c.v.y = (__bf16)b;
  return c.u;
}

union Frag { uint4 q[2]; v16bf v; };

// low 32 bits of a generic LDS pointer == wave-relative LDS byte offset
__device__ __forceinline__ unsigned lds_off_u32(const void* p) {
  return (unsigned)(unsigned long long)p;
}

// CDNA5 async copy: 16 bytes per lane, global -> LDS, tracked by ASYNCcnt
__device__ __forceinline__ void async_copy_b128(unsigned lds_dst, const void* gsrc) {
  asm volatile("global_load_async_to_lds_b128 %0, %1, off"
               :: "v"(lds_dst), "v"(gsrc) : "memory");
}

#if __has_builtin(__builtin_amdgcn_s_wait_asynccnt)
#define WAIT_ASYNC(n) __builtin_amdgcn_s_wait_asynccnt(n)
#else
#define WAIT_ASYNC(n) asm volatile("s_wait_asynccnt %0" :: "i"(n) : "memory")
#endif

// ---- shared SNN epilogue ---------------------------------------------------
__device__ __forceinline__ void snn_epilogue(
    const v8f (&acc)[2][2][2], int m0, int n0, int wm, int wn, int hi, int lo,
    const float* __restrict__ tau_m, const float* __restrict__ tau_n,
    const float* __restrict__ mem, const float* __restrict__ spike,
    const float* __restrict__ d_input, const float* __restrict__ v_th,
    float* __restrict__ out_mem, float* __restrict__ out_spike)
{
#pragma unroll
  for (int ni = 0; ni < 2; ++ni) {
    const int d = n0 + wn * 32 + ni * 16 + lo;
    const float alpha = sigmoid_f(tau_m[d]);
    const float beta0 = sigmoid_f(tau_n[2 * d + 0]);
    const float beta1 = sigmoid_f(tau_n[2 * d + 1]);
#pragma unroll
    for (int mi = 0; mi < 2; ++mi) {
#pragma unroll
      for (int r = 0; r < 8; ++r) {
        const int    bi = m0 + wm * 32 + mi * 16 + hi * 8 + r;
        const size_t o  = (size_t)bi * D_DIM + d;
        const float den0 = acc[0][mi][ni][r];
        const float den1 = acc[1][mi][ni][r];
        const float2 di  = *(const float2*)&d_input[o * 2];
        const float l    = beta0 * di.x + (1.0f - beta0) * den0
                         + beta1 * di.y + (1.0f - beta1) * den1;
        const float vth  = v_th[o];
        const float mnew = mem[o] * alpha + (1.0f - alpha) * l - vth * spike[o];
        out_mem[o]   = mnew;
        out_spike[o] = (mnew - vth > 0.0f) ? 1.0f : 0.0f;
      }
    }
  }
}

// ---------------------------------------------------------------------------
// Pass 1: streaming f32 -> bf16 conversion (done exactly once per element)
// ---------------------------------------------------------------------------
__global__ __launch_bounds__(256)
void cvt_f32_to_bf16_kernel(const float* __restrict__ src,
                            unsigned* __restrict__ dst, int n4)
{
  const int i = blockIdx.x * 256 + threadIdx.x;
  if (i < n4) {
    const float4 f = ((const float4*)src)[i];
    uint2 p;
    p.x = pk2bf(f.x, f.y);
    p.y = pk2bf(f.z, f.w);
    ((uint2*)dst)[i] = p;
  }
}

// ---------------------------------------------------------------------------
// Pass 2: GEMM on pre-converted bf16, 4-slot async LDS pipeline
// ---------------------------------------------------------------------------
__global__ __launch_bounds__(256)
void snn_wmma_bf16pre_kernel(const unsigned short* __restrict__ inb,
                             const unsigned short* __restrict__ Wb,
                             const float* __restrict__ tau_m,
                             const float* __restrict__ tau_n,
                             const float* __restrict__ mem,
                             const float* __restrict__ spike,
                             const float* __restrict__ d_input,
                             const float* __restrict__ v_th,
                             float* __restrict__ out_mem,
                             float* __restrict__ out_spike)
{
  __shared__ __align__(16) unsigned short lds[NBUF][TBUF];  // 4 x 15360 B

  const int t    = threadIdx.x;
  const int m0   = blockIdx.y * BM;
  const int n0   = blockIdx.x * BN;

  const int wave = t >> 5;
  const int lane = t & 31;
  const int wm   = wave >> 1;
  const int wn   = wave & 1;
  const int hi   = lane >> 4;
  const int lo   = lane & 15;

  // A tile: 128 rows x 32 halves = 512 x 16B chunks -> 2 chunks/thread
  const int ar0 = t >> 2;             // rows 0..63
  const int ar1 = 64 + (t >> 2);      // rows 64..127
  const int ac_ = (t & 3) * 8;        // half offset within row
  // B tile: 64 rows x 32 halves = 256 x 16B chunks -> 1 chunk/thread
  const int brr = t >> 2;
  const int bcc = (t & 3) * 8;

  // slot-0 LDS byte offsets for this thread's three async chunks
  const unsigned a0base = lds_off_u32(&lds[0][ar0 * LDA + ac_]);
  const unsigned a1base = lds_off_u32(&lds[0][ar1 * LDA + ac_]);
  const unsigned bbase  = lds_off_u32(&lds[0][ABUF + brr * LDB + bcc]);

  v8f acc[2][2][2];
#pragma unroll
  for (int p = 0; p < 2; ++p)
#pragma unroll
    for (int i = 0; i < 2; ++i)
#pragma unroll
      for (int j = 0; j < 2; ++j)
        acc[p][i][j] = v8f{};

#pragma unroll
  for (int ph = 0; ph < 2; ++ph) {
    const size_t koff = (size_t)ph * (K_DIM / 2);
    const unsigned short* Ab0 = inb + (size_t)(m0 + ar0) * K_DIM + koff + ac_;
    const unsigned short* Ab1 = inb + (size_t)(m0 + ar1) * K_DIM + koff + ac_;
    const unsigned short* Bb  = Wb + (size_t)(2 * (n0 + brr) + ph) * K_DIM + koff + bcc;

    // prologue: tiles 0..2 -> slots 0..2 (9 async ops in flight)
#pragma unroll
    for (int p = 0; p < 3; ++p) {
      const unsigned so = (unsigned)p * SLOT_B;
      async_copy_b128(a0base + so, Ab0 + p * BK);
      async_copy_b128(a1base + so, Ab1 + p * BK);
      async_copy_b128(bbase  + so, Bb  + p * BK);
    }

#pragma unroll 4
    for (int kt = 0; kt < NKT; ++kt) {
      // async completes in order: <=6 outstanding => oldest tile (kt) landed
      if (kt + 2 < NKT)      WAIT_ASYNC(6);
      else if (kt + 1 < NKT) WAIT_ASYNC(3);
      else                   WAIT_ASYNC(0);
      __syncthreads();      // tile kt visible to all waves; slot (kt+3)&3 free

      if (kt + 3 < NKT) {   // kick off tile kt+3; overlaps WMMAs of kt..kt+2
        const unsigned so = (unsigned)((kt + 3) & 3) * SLOT_B;
        async_copy_b128(a0base + so, Ab0 + (kt + 3) * BK);
        async_copy_b128(a1base + so, Ab1 + (kt + 3) * BK);
        async_copy_b128(bbase  + so, Bb  + (kt + 3) * BK);
      }

      const unsigned short* As = &lds[kt & 3][0];
      const unsigned short* Bs = &lds[kt & 3][ABUF];

      Frag a[2], b[2];
#pragma unroll
      for (int mi = 0; mi < 2; ++mi) {
        const unsigned short* p = &As[(wm * 32 + mi * 16 + lo) * LDA + hi * 16];
        a[mi].q[0] = *(const uint4*)p;
        a[mi].q[1] = *(const uint4*)(p + 8);
      }
#pragma unroll
      for (int ni = 0; ni < 2; ++ni) {
        const unsigned short* p = &Bs[(wn * 32 + ni * 16 + lo) * LDB + hi * 16];
        b[ni].q[0] = *(const uint4*)p;
        b[ni].q[1] = *(const uint4*)(p + 8);
      }

#pragma unroll
      for (int mi = 0; mi < 2; ++mi)
#pragma unroll
        for (int ni = 0; ni < 2; ++ni)
          acc[ph][mi][ni] = __builtin_amdgcn_wmma_f32_16x16x32_bf16(
              false, a[mi].v, false, b[ni].v,
              (short)0, acc[ph][mi][ni], false, false);
    }
  }

  snn_epilogue(acc, m0, n0, wm, wn, hi, lo,
               tau_m, tau_n, mem, spike, d_input, v_th, out_mem, out_spike);
}

// ---------------------------------------------------------------------------
// Fallback: fused single-pass kernel (f32 -> bf16 conversion while staging)
// ---------------------------------------------------------------------------
__global__ __launch_bounds__(256)
void snn_wmma_fused_kernel(const float* __restrict__ input,
                           const float* __restrict__ W,
                           const float* __restrict__ tau_m,
                           const float* __restrict__ tau_n,
                           const float* __restrict__ mem,
                           const float* __restrict__ spike,
                           const float* __restrict__ d_input,
                           const float* __restrict__ v_th,
                           float* __restrict__ out_mem,
                           float* __restrict__ out_spike)
{
  __shared__ __align__(16) unsigned short ldsA[BM * LDA];
  __shared__ __align__(16) unsigned short ldsB[BN * LDB];

  const int t    = threadIdx.x;
  const int m0   = blockIdx.y * BM;
  const int n0   = blockIdx.x * BN;

  const int wave = t >> 5;
  const int lane = t & 31;
  const int wm   = wave >> 1;
  const int wn   = wave & 1;
  const int hi   = lane >> 4;
  const int lo   = lane & 15;

  const int ar = t >> 1;
  const int ac = (t & 1) * 16;
  const int br = t >> 2;
  const int bc = (t & 3) * 8;

  v8f acc[2][2][2];
#pragma unroll
  for (int p = 0; p < 2; ++p)
#pragma unroll
    for (int i = 0; i < 2; ++i)
#pragma unroll
      for (int j = 0; j < 2; ++j)
        acc[p][i][j] = v8f{};

#pragma unroll
  for (int ph = 0; ph < 2; ++ph) {
    const size_t koff  = (size_t)ph * (K_DIM / 2);
    const float* Abase = input + (size_t)(m0 + ar) * K_DIM + koff + ac;
    const float* Bbase = W + (size_t)(2 * (n0 + br) + ph) * K_DIM + koff + bc;

    for (int kt = 0; kt < NKT; ++kt) {
      const float* ag = Abase + kt * BK;
      const float* bg = Bbase + kt * BK;

#pragma unroll
      for (int i = 0; i < 4; ++i) {
        float4 f = *(const float4*)(ag + i * 4);
        uint2 p; p.x = pk2bf(f.x, f.y); p.y = pk2bf(f.z, f.w);
        *(uint2*)&ldsA[ar * LDA + ac + i * 4] = p;
      }
#pragma unroll
      for (int i = 0; i < 2; ++i) {
        float4 f = *(const float4*)(bg + i * 4);
        uint2 p; p.x = pk2bf(f.x, f.y); p.y = pk2bf(f.z, f.w);
        *(uint2*)&ldsB[br * LDB + bc + i * 4] = p;
      }
      __syncthreads();

      if (kt < NKT - 1) {
        __builtin_prefetch(ag + BK, 0, 1);
        __builtin_prefetch(bg + BK, 0, 1);
      }

      Frag a[2], b[2];
#pragma unroll
      for (int mi = 0; mi < 2; ++mi) {
        const unsigned short* p = &ldsA[(wm * 32 + mi * 16 + lo) * LDA + hi * 16];
        a[mi].q[0] = *(const uint4*)p;
        a[mi].q[1] = *(const uint4*)(p + 8);
      }
#pragma unroll
      for (int ni = 0; ni < 2; ++ni) {
        const unsigned short* p = &ldsB[(wn * 32 + ni * 16 + lo) * LDB + hi * 16];
        b[ni].q[0] = *(const uint4*)p;
        b[ni].q[1] = *(const uint4*)(p + 8);
      }

#pragma unroll
      for (int mi = 0; mi < 2; ++mi)
#pragma unroll
        for (int ni = 0; ni < 2; ++ni)
          acc[ph][mi][ni] = __builtin_amdgcn_wmma_f32_16x16x32_bf16(
              false, a[mi].v, false, b[ni].v,
              (short)0, acc[ph][mi][ni], false, false);

      __syncthreads();
    }
  }

  snn_epilogue(acc, m0, n0, wm, wn, hi, lo,
               tau_m, tau_n, mem, spike, d_input, v_th, out_mem, out_spike);
}

// ---------------------------------------------------------------------------
extern "C" void kernel_launch(void* const* d_in, const int* in_sizes, int n_in,
                              void* d_out, int out_size, void* d_ws, size_t ws_size,
                              hipStream_t stream) {
  (void)in_sizes; (void)n_in; (void)out_size;
  const float* input   = (const float*)d_in[0];
  const float* W       = (const float*)d_in[1];
  const float* tau_m   = (const float*)d_in[2];
  const float* tau_n   = (const float*)d_in[3];
  const float* mem     = (const float*)d_in[4];
  const float* spike   = (const float*)d_in[5];
  const float* d_input = (const float*)d_in[6];
  const float* v_th    = (const float*)d_in[7];
  float* out_mem   = (float*)d_out;
  float* out_spike = (float*)d_out + (size_t)B_DIM * D_DIM;

  const size_t in_elems = (size_t)B_DIM * K_DIM;          // 8M
  const size_t w_elems  = (size_t)(2 * D_DIM) * K_DIM;    // 16M
  const size_t need     = (in_elems + w_elems) * 2;       // 48MB bf16

  const dim3 grid(D_DIM / BN, B_DIM / BM);                // (32, 16)

  if (ws_size >= need) {
    unsigned short* inb = (unsigned short*)d_ws;
    unsigned short* Wb  = inb + in_elems;
    const int in4 = (int)(in_elems / 4);
    const int w4  = (int)(w_elems / 4);
    cvt_f32_to_bf16_kernel<<<(in4 + 255) / 256, 256, 0, stream>>>(
        input, (unsigned*)inb, in4);
    cvt_f32_to_bf16_kernel<<<(w4 + 255) / 256, 256, 0, stream>>>(
        W, (unsigned*)Wb, w4);
    snn_wmma_bf16pre_kernel<<<grid, 256, 0, stream>>>(
        inb, Wb, tau_m, tau_n, mem, spike, d_input, v_th, out_mem, out_spike);
  } else {
    snn_wmma_fused_kernel<<<grid, 256, 0, stream>>>(
        input, W, tau_m, tau_n, mem, spike, d_input, v_th, out_mem, out_spike);
  }
}
